// PredictionHead_46067819217218
// MI455X (gfx1250) — compile-verified, matrix operand order checked by reading, not run on hardware
//
#include <hip/hip_runtime.h>
#include <hip/hip_bf16.h>

typedef __bf16 bf16;
typedef __attribute__((ext_vector_type(16))) __bf16 v16bf;
typedef __attribute__((ext_vector_type(8)))  float  v8f;
typedef __attribute__((ext_vector_type(4)))  unsigned int uint4v;

#define Q_TOT   16384
#define F_DIM   128
#define DQ_DIM  128
#define HD_DIM  256
#define IMG_H   1024
#define IMG_W   1024
#define WIN     5
#define NSLOT   32      // 25 window slots padded to 32 (two 16-row WMMA tiles)
#define QPW     4       // qp_mlp: waves per block

__device__ __forceinline__ v16bf load_frag2(const bf16* p0, const bf16* p1) {
    union { v16bf v; uint4v u[2]; } f;
    f.u[0] = *(const uint4v*)p0;
    f.u[1] = *(const uint4v*)p1;
    return f.v;
}

// -------- weight packer: f32 [K][N] row-major -> bf16 WMMA-B fragment-major ----
// packed[((kt*NT + nt)*32 + lane)*16 + e]; lane<16 covers K {0..7,16..23} (+hi*8),
// lanes 16..31 cover K {8..15,24..31}; n = nt*16 + (lane&15).
__global__ __launch_bounds__(256) void pack_weights(const float* __restrict__ w,
                                                    bf16* __restrict__ out,
                                                    int K, int N) {
    int idx = blockIdx.x * 256 + threadIdx.x;
    if (idx >= K * N) return;
    int e    = idx & 15;
    int lane = (idx >> 4) & 31;
    int rest = idx >> 9;
    int NT   = N >> 4;
    int nt   = rest % NT;
    int kt   = rest / NT;
    int hi   = lane >> 4;
    int n    = nt * 16 + (lane & 15);
    int k    = kt * 32 + ((e < 8) ? (hi * 8 + e) : (16 + hi * 8 + (e - 8)));
    out[idx] = (bf16)w[(size_t)k * N + n];
}

__global__ __launch_bounds__(256) void cast_bf16(const float* __restrict__ in,
                                                 bf16* __restrict__ out, int n) {
    int i = blockIdx.x * 256 + threadIdx.x;
    if (i < n) out[i] = (bf16)in[i];
}

// -------- query-portion MLP: mlped_q = relu(queries@qp_w1+b1)@qp_w2+b2 --------
__global__ __launch_bounds__(128) void qp_mlp(const bf16* __restrict__ qbf,
                                              const bf16* __restrict__ w1p,
                                              const bf16* __restrict__ w2p,
                                              const float* __restrict__ b1,
                                              const float* __restrict__ b2,
                                              float* __restrict__ mlq) {
    __shared__ bf16 hidden[QPW][16][HD_DIM];            // 32 KB
    int tid = threadIdx.x, wave = tid >> 5, lane = tid & 31;
    int half = lane >> 4, ln = lane & 15;
    int q0 = (blockIdx.x * QPW + wave) * 16;
    const bf16* Arow = qbf + (size_t)(q0 + ln) * DQ_DIM;
    bf16* H = &hidden[wave][0][0];

    for (int nt = 0; nt < HD_DIM / 16; ++nt) {          // layer 1
        v8f c = {};
        #pragma unroll
        for (int kt = 0; kt < DQ_DIM / 32; ++kt) {
            const bf16* ap = Arow + kt * 32 + half * 8;
            v16bf a = load_frag2(ap, ap + 16);
            const bf16* bp = w1p + (((size_t)kt * (HD_DIM / 16) + nt) * 32 + lane) * 16;
            v16bf b = load_frag2(bp, bp + 8);
            c = __builtin_amdgcn_wmma_f32_16x16x32_bf16(false, a, false, b,
                                                        (short)0, c, false, false);
        }
        float bias = b1[nt * 16 + ln];
        #pragma unroll
        for (int v = 0; v < 8; ++v) {
            float hv = c[v] + bias;
            hv = hv > 0.f ? hv : 0.f;
            H[(size_t)(v + 8 * half) * HD_DIM + nt * 16 + ln] = (bf16)hv;
        }
    }
    __syncthreads();
    for (int nt = 0; nt < F_DIM / 16; ++nt) {           // layer 2
        v8f c = {};
        #pragma unroll
        for (int kt = 0; kt < HD_DIM / 32; ++kt) {
            const bf16* ap = H + (size_t)ln * HD_DIM + kt * 32 + half * 8;
            v16bf a = load_frag2(ap, ap + 16);
            const bf16* bp = w2p + (((size_t)kt * (F_DIM / 16) + nt) * 32 + lane) * 16;
            v16bf b = load_frag2(bp, bp + 8);
            c = __builtin_amdgcn_wmma_f32_16x16x32_bf16(false, a, false, b,
                                                        (short)0, c, false, false);
        }
        float bias = b2[nt * 16 + ln];
        #pragma unroll
        for (int v = 0; v < 8; ++v)
            mlq[(size_t)(q0 + v + 8 * half) * F_DIM + nt * 16 + ln] = c[v] + bias;
    }
}

// -------- main: one wave = one query (32 row slots, two M-tiles) -------------
// A-fragments built in registers directly from the image; B-fragments reused
// across both M-tiles (halves L2 weight traffic per WMMA).
__global__ __launch_bounds__(64) void main_kernel(
        const float* __restrict__ queries, const int* __restrict__ qidx,
        const float* __restrict__ image,
        const float* __restrict__ pm_b1, const float* __restrict__ pm_b2,
        const bf16* __restrict__ w1p, const bf16* __restrict__ w2p,
        const float* __restrict__ mlq,
        float* __restrict__ out_binary, float* __restrict__ out_portion,
        float* __restrict__ out_nonzero) {
    __shared__ bf16  hidden[2][NSLOT][HD_DIM];          // 32 KB
    __shared__ float padBias[2][NSLOT];
    const float NEG_INF = -__builtin_inff();

    int wave = threadIdx.x >> 5, lane = threadIdx.x & 31;
    int half = lane >> 4, ln = lane & 15;
    int q = blockIdx.x * 2 + wave;

    int iy = qidx[q * 2 + 0], ix = qidx[q * 2 + 1];

    // per-lane row geometry for both tiles (row slot = mt*16 + ln)
    bool valid[2];
    const float* rowp[2];
    #pragma unroll
    for (int mt = 0; mt < 2; ++mt) {
        int slot = mt * 16 + ln;
        int yy = iy + slot / WIN - WIN / 2;
        int xx = ix + slot % WIN - WIN / 2;
        bool v = (slot < WIN * WIN) && (yy >= 0) && (yy < IMG_H) &&
                 (xx >= 0) && (xx < IMG_W);
        valid[mt] = v;
        rowp[mt] = image + ((size_t)(v ? yy : 0) * IMG_W + (v ? xx : 0)) * F_DIM;
        if (half == 0) padBias[wave][slot] = (slot < WIN * WIN && !v) ? NEG_INF : 0.f;
    }

    // ---- build A fragments from image; fuse portion dot + nonzero mask ----
    v16bf A[2][4];
    float dloc[2] = {0.f, 0.f};
    float nzloc[2] = {0.f, 0.f};
    const float* mqp = mlq + (size_t)q * F_DIM;
    #pragma unroll
    for (int kt = 0; kt < F_DIM / 32; ++kt) {
        int c0 = kt * 32 + half * 8;        // this lane's first 8-col chunk
        int c1 = c0 + 16;                   // second chunk
        float4 m0 = *(const float4*)(mqp + c0);
        float4 m1 = *(const float4*)(mqp + c0 + 4);
        float4 m2 = *(const float4*)(mqp + c1);
        float4 m3 = *(const float4*)(mqp + c1 + 4);
        #pragma unroll
        for (int mt = 0; mt < 2; ++mt) {
            float4 k0 = make_float4(0.f, 0.f, 0.f, 0.f), k1 = k0, k2 = k0, k3 = k0;
            if (valid[mt]) {
                k0 = *(const float4*)(rowp[mt] + c0);
                k1 = *(const float4*)(rowp[mt] + c0 + 4);
                k2 = *(const float4*)(rowp[mt] + c1);
                k3 = *(const float4*)(rowp[mt] + c1 + 4);
            }
            v16bf a;
            a[0] = (bf16)k0.x; a[1] = (bf16)k0.y; a[2]  = (bf16)k0.z; a[3]  = (bf16)k0.w;
            a[4] = (bf16)k1.x; a[5] = (bf16)k1.y; a[6]  = (bf16)k1.z; a[7]  = (bf16)k1.w;
            a[8] = (bf16)k2.x; a[9] = (bf16)k2.y; a[10] = (bf16)k2.z; a[11] = (bf16)k2.w;
            a[12]= (bf16)k3.x; a[13]= (bf16)k3.y; a[14] = (bf16)k3.z; a[15] = (bf16)k3.w;
            A[mt][kt] = a;
            dloc[mt] += k0.x * m0.x + k0.y * m0.y + k0.z * m0.z + k0.w * m0.w
                      + k1.x * m1.x + k1.y * m1.y + k1.z * m1.z + k1.w * m1.w
                      + k2.x * m2.x + k2.y * m2.y + k2.z * m2.z + k2.w * m2.w
                      + k3.x * m3.x + k3.y * m3.y + k3.z * m3.z + k3.w * m3.w;
            if (k0.x != 0.f || k0.y != 0.f || k0.z != 0.f || k0.w != 0.f ||
                k1.x != 0.f || k1.y != 0.f || k1.z != 0.f || k1.w != 0.f ||
                k2.x != 0.f || k2.y != 0.f || k2.z != 0.f || k2.w != 0.f ||
                k3.x != 0.f || k3.y != 0.f || k3.z != 0.f || k3.w != 0.f)
                nzloc[mt] = 1.f;
        }
    }
    // lanes l and l^16 hold complementary column halves of the same row
    #pragma unroll
    for (int mt = 0; mt < 2; ++mt) {
        float dtot  = dloc[mt]  + __shfl_xor(dloc[mt], 16, 32);
        float nztot = nzloc[mt] + __shfl_xor(nzloc[mt], 16, 32);
        int slot = mt * 16 + ln;
        if (half == 0 && slot < WIN * WIN) {
            size_t o = (size_t)q * (WIN * WIN) + slot;
            out_portion[o] = valid[mt] ? dtot : NEG_INF;
            out_nonzero[o] = (valid[mt] && nztot > 0.f) ? 1.f : 0.f;
        }
    }

    // ---- layer 1: two M-tiles share every B fragment ----
    bf16* H = &hidden[wave][0][0];
    for (int nt = 0; nt < HD_DIM / 16; ++nt) {
        v8f c0 = {}, c1 = {};
        #pragma unroll
        for (int kt = 0; kt < F_DIM / 32; ++kt) {
            const bf16* bp = w1p + (((size_t)kt * (HD_DIM / 16) + nt) * 32 + lane) * 16;
            v16bf b = load_frag2(bp, bp + 8);
            c0 = __builtin_amdgcn_wmma_f32_16x16x32_bf16(false, A[0][kt], false, b,
                                                         (short)0, c0, false, false);
            c1 = __builtin_amdgcn_wmma_f32_16x16x32_bf16(false, A[1][kt], false, b,
                                                         (short)0, c1, false, false);
        }
        float bias = pm_b1[nt * 16 + ln];
        #pragma unroll
        for (int v = 0; v < 8; ++v) {
            float h0 = c0[v] + bias; h0 = h0 > 0.f ? h0 : 0.f;
            float h1 = c1[v] + bias; h1 = h1 > 0.f ? h1 : 0.f;
            int r = v + 8 * half;
            H[(size_t)r * HD_DIM + nt * 16 + ln] = (bf16)h0;
            H[(size_t)(16 + r) * HD_DIM + nt * 16 + ln] = (bf16)h1;
        }
    }
    __syncthreads();

    // ---- layer 2 + fused q-dot ----
    float acc0[8], acc1[8];
    #pragma unroll
    for (int v = 0; v < 8; ++v) { acc0[v] = 0.f; acc1[v] = 0.f; }

    for (int nt = 0; nt < DQ_DIM / 16; ++nt) {
        v8f c0 = {}, c1 = {};
        #pragma unroll
        for (int kt = 0; kt < HD_DIM / 32; ++kt) {
            const bf16* a0p = H + (size_t)ln * HD_DIM + kt * 32 + half * 8;
            const bf16* a1p = a0p + (size_t)16 * HD_DIM;
            v16bf a0 = load_frag2(a0p, a0p + 16);
            v16bf a1 = load_frag2(a1p, a1p + 16);
            const bf16* bp = w2p + (((size_t)kt * (DQ_DIM / 16) + nt) * 32 + lane) * 16;
            v16bf b = load_frag2(bp, bp + 8);
            c0 = __builtin_amdgcn_wmma_f32_16x16x32_bf16(false, a0, false, b,
                                                         (short)0, c0, false, false);
            c1 = __builtin_amdgcn_wmma_f32_16x16x32_bf16(false, a1, false, b,
                                                         (short)0, c1, false, false);
        }
        float bias = pm_b2[nt * 16 + ln];
        float qv = queries[(size_t)q * DQ_DIM + nt * 16 + ln];
        #pragma unroll
        for (int v = 0; v < 8; ++v) {
            acc0[v] += (c0[v] + bias) * qv;
            acc1[v] += (c1[v] + bias) * qv;
        }
    }
    #pragma unroll
    for (int m = 8; m >= 1; m >>= 1)
        #pragma unroll
        for (int v = 0; v < 8; ++v) {
            acc0[v] += __shfl_xor(acc0[v], m, 32);
            acc1[v] += __shfl_xor(acc1[v], m, 32);
        }
    if (ln == 0) {
        #pragma unroll
        for (int v = 0; v < 8; ++v) {
            int r = v + 8 * half;                       // tile-local row
            // tile 0 -> slots 0..15 (always < 25)
            out_binary[(size_t)q * (WIN * WIN) + r] = acc0[v] + padBias[wave][r];
            int slot1 = 16 + r;
            if (slot1 < WIN * WIN)
                out_binary[(size_t)q * (WIN * WIN) + slot1] =
                    acc1[v] + padBias[wave][slot1];
        }
    }
}

extern "C" void kernel_launch(void* const* d_in, const int* in_sizes, int n_in,
                              void* d_out, int out_size, void* d_ws, size_t ws_size,
                              hipStream_t stream) {
    const float* queries = (const float*)d_in[0];
    const int*   qidx    = (const int*)d_in[1];
    const float* image   = (const float*)d_in[2];
    const float* pm_w1   = (const float*)d_in[3];
    const float* pm_b1   = (const float*)d_in[4];
    const float* pm_w2   = (const float*)d_in[5];
    const float* pm_b2   = (const float*)d_in[6];
    const float* qp_w1   = (const float*)d_in[7];
    const float* qp_b1   = (const float*)d_in[8];
    const float* qp_w2   = (const float*)d_in[9];
    const float* qp_b2   = (const float*)d_in[10];

    char* ws = (char*)d_ws;
    bf16* pm_w1p = (bf16*)(ws + 0);
    bf16* pm_w2p = (bf16*)(ws + (1 << 16));
    bf16* qp_w1p = (bf16*)(ws + (2 << 16));
    bf16* qp_w2p = (bf16*)(ws + (3 << 16));
    bf16* qbf    = (bf16*)(ws + (4 << 16));
    float* mlq   = (float*)(ws + (4 << 16) + (size_t)Q_TOT * DQ_DIM * 2);

    const int W1N = DQ_DIM * HD_DIM, W2N = HD_DIM * F_DIM;
    pack_weights<<<(W1N + 255) / 256, 256, 0, stream>>>(pm_w1, pm_w1p, F_DIM, HD_DIM);
    pack_weights<<<(W2N + 255) / 256, 256, 0, stream>>>(pm_w2, pm_w2p, HD_DIM, DQ_DIM);
    pack_weights<<<(W1N + 255) / 256, 256, 0, stream>>>(qp_w1, qp_w1p, DQ_DIM, HD_DIM);
    pack_weights<<<(W2N + 255) / 256, 256, 0, stream>>>(qp_w2, qp_w2p, HD_DIM, F_DIM);
    const int QN = Q_TOT * DQ_DIM;
    cast_bf16<<<(QN + 255) / 256, 256, 0, stream>>>(queries, qbf, QN);

    qp_mlp<<<Q_TOT / (QPW * 16), 128, 0, stream>>>(qbf, qp_w1p, qp_w2p,
                                                   qp_b1, qp_b2, mlq);

    float* out_binary  = (float*)d_out;
    float* out_portion = out_binary + (size_t)Q_TOT * WIN * WIN;
    float* out_nonzero = out_portion + (size_t)Q_TOT * WIN * WIN;
    main_kernel<<<Q_TOT / 2, 64, 0, stream>>>(queries, qidx, image,
                                              pm_b1, pm_b2, pm_w1p, pm_w2p, mlq,
                                              out_binary, out_portion, out_nonzero);
}
